// Decoder_70720931496166
// MI455X (gfx1250) — matrix operand order repaired and checked by reading, not compiled, not run on hardware
//
#include <hip/hip_runtime.h>

// ---------------------------------------------------------------------------
// Tacotron decoder + CBHG postnet for gfx1250 (MI455X), wave32 + WMMA f16.
// v2: all GEMM A-operands (activations / conv inputs / GRU state mirrors) are
// kept in f16 so WMMA fragments are raw b128 loads with no per-load cvt VALU.
// f32 masters retained only for elementwise math and final outputs.
// ---------------------------------------------------------------------------

typedef __attribute__((ext_vector_type(16))) _Float16 v16h;
typedef __attribute__((ext_vector_type(8)))  _Float16 v8h;
typedef __attribute__((ext_vector_type(8)))  float    v8f;

#define DEV __device__ __forceinline__

DEV v8f wmma_f16(v16h a, v16h b, v8f c) {
  // (neg_a, A, neg_b, B, c_mod, C, reuse_a, reuse_b)
  return __builtin_amdgcn_wmma_f32_16x16x32_f16(false, a, false, b, (short)0, c,
                                                false, false);
}

DEV float sigmoidf_(float x) { return 1.f / (1.f + __expf(-x)); }

// ---- fragment loaders -----------------------------------------------------
// A (16x32 f16): lane l holds row M=l%16; element j -> K = (l/16)*8 + (j/8)*16 + j%8
// B (32x16 f16 from Wt[N][K] row-major): lane l -> N=l%16; element j -> K = (l/16)*8 + (j/8)*16 + j%8
// C/D (16x16 f32): lane l, reg r -> (M=(l/16)*8+r, N=l%16)

struct Asrc16 {            // element(m,k) = k<klen0 ? p0[m*s0+k] : p1[m*s1+k-klen0]
  const _Float16* p0; long s0; int klen0;
  const _Float16* p1; long s1;
};

// 8-half contiguous chunk; callers guarantee klen0 % 8 == 0 (no straddle).
DEV v8h s_chunk(const Asrc16& s, int m, int k, int Kt) {
  const _Float16* src; long off;
  if (k >= s.klen0) { src = s.p1; off = (long)m * s.s1 + (k - s.klen0); }
  else              { src = s.p0; off = (long)m * s.s0 + k; }
  if (k + 8 <= Kt) return *(const v8h*)(src + off);   // 16B aligned by layout
  v8h r{};
#pragma unroll
  for (int j = 0; j < 8; ++j)
    if (k + j < Kt) r[j] = src[off + j];
  return r;
}

DEV v16h frag_a16(const Asrc16& s, int Kt, int row0, int k0) {
  int lane = threadIdx.x & 31;
  int m = row0 + (lane & 15);
  int kb = k0 + ((lane >> 4) << 3);
  v8h lo = s_chunk(s, m, kb, Kt);
  v8h hi = s_chunk(s, m, kb + 16, Kt);
  v16h a;
#pragma unroll
  for (int j = 0; j < 8; ++j) { a[j] = lo[j]; a[8 + j] = hi[j]; }
  return a;
}

DEV v16h frag_b16(const _Float16* Wt, int N, int K, int n0, int k0) {
  int lane = threadIdx.x & 31;
  int n = n0 + (lane & 15);
  int kb = k0 + ((lane >> 4) << 3);
  v16h b;
  const _Float16* bp = Wt + (long)n * K + kb;
  if (n < N && kb + 24 <= K) {       // fast: two b128 loads
    v8h lo = *(const v8h*)bp;
    v8h hi = *(const v8h*)(bp + 16);
#pragma unroll
    for (int j = 0; j < 8; ++j) { b[j] = lo[j]; b[8 + j] = hi[j]; }
  } else {                            // ragged N (1025) / ragged K tail
#pragma unroll
    for (int j = 0; j < 16; ++j) {
      int k = kb + j + (j & 8);
      b[j] = (n < N && k < K) ? Wt[(long)n * K + k] : (_Float16)0.f;
    }
  }
  return b;
}

// ---------------------------------------------------------------------------
// Generic batched GEMM kernel: C[M,N] = epilogue(A[M,K] @ Wt[N,K]^T)
// MODE 0: plain f16 A; MODE 1: im2col conv (f16 X); MODE 2: conv over
// maxpool(l,l+1); MODE 3: concat(A,A2) along K. M is always a multiple of 16.
// ---------------------------------------------------------------------------
struct GemmP {
  const _Float16* A; long lda;
  const _Float16* A2; long lda2; int ksplit;   // mode 3 (ksplit % 8 == 0)
  const _Float16* X; int Lc, Cin, kw, pad;     // conv modes (Cin % 8 == 0)
  const _Float16* Wt;
  const float* bias; const float* scale; const float* shift;
  const float* res; long ldres;
  float* C; long ldc;            // optional f32 output
  _Float16* C16; long ldc16;     // optional f16 output
  int M, N, K, act;              // act: 0 none, 1 relu, 2 sigmoid
};

template <int MODE> DEV v8h a_chunk(const GemmP& p, int m, int k) {
  if (MODE == 0 || MODE == 3) {
    const _Float16* src; long off;
    if (MODE == 3 && k >= p.ksplit) { src = p.A2; off = (long)m * p.lda2 + (k - p.ksplit); }
    else                            { src = p.A;  off = (long)m * p.lda + k; }
    if (k + 8 <= p.K) return *(const v8h*)(src + off);
    v8h r{};
#pragma unroll
    for (int j = 0; j < 8; ++j)
      if (k + j < p.K) r[j] = src[off + j];
    return r;
  } else {
    v8h r{};
    if (k + 8 > p.K) return r;        // conv K always multiple of 32
    int tap = k / p.Cin;
    int c = k - tap * p.Cin;          // c % 8 == 0 since Cin % 8 == 0
    int b = m / p.Lc;
    int l = (m - b * p.Lc) + tap - p.pad;
    if (l < 0 || l >= p.Lc) return r; // conv zero padding
    const _Float16* xp = p.X + ((long)(b * p.Lc + l)) * p.Cin + c;
    v8h v = *(const v8h*)xp;
    if (MODE == 2 && (l + 1) < p.Lc) { // fused width-2 maxpool
      v8h w = *(const v8h*)(xp + p.Cin);
#pragma unroll
      for (int j = 0; j < 8; ++j) v[j] = (w[j] > v[j]) ? w[j] : v[j];
    }
    return v;
  }
}

template <int MODE> DEV v16h frag_a_g(const GemmP& p, int row0, int k0) {
  int lane = threadIdx.x & 31;
  int m = row0 + (lane & 15);
  int kb = k0 + ((lane >> 4) << 3);
  v8h lo = a_chunk<MODE>(p, m, kb);
  v8h hi = a_chunk<MODE>(p, m, kb + 16);
  v16h a;
#pragma unroll
  for (int j = 0; j < 8; ++j) { a[j] = lo[j]; a[8 + j] = hi[j]; }
  return a;
}

template <int MODE>
__global__ void __launch_bounds__(128) k_gemm(GemmP p) {
  int ntw = (p.N + 15) >> 4;
  int total = ((p.M + 15) >> 4) * ntw;
  int wave = blockIdx.x * (blockDim.x >> 5) + (threadIdx.x >> 5);
  int step = gridDim.x * (blockDim.x >> 5);
  for (int u = wave; u < total; u += step) {
    int mt = u / ntw, nt = u - mt * ntw;
    int row0 = mt << 4, n0 = nt << 4;
    v8f acc = {};
    for (int k0 = 0; k0 < p.K; k0 += 32) {
      if constexpr (MODE == 0) {   // hint next A chunk -> global_prefetch_b8
        int lane = threadIdx.x & 31;
        const _Float16* pf = p.A + (long)(row0 + (lane & 15)) * p.lda + k0 + 32;
        __builtin_prefetch(pf, 0, 1);
      }
      v16h a = frag_a_g<MODE>(p, row0, k0);
      v16h b = frag_b16(p.Wt, p.N, p.K, n0, k0);
      acc = wmma_f16(a, b, acc);
    }
    int lane = threadIdx.x & 31;
    int n = n0 + (lane & 15);
    if (n < p.N) {
      float sc = p.scale ? p.scale[n] : 1.f;
      float sh = p.shift ? p.shift[n] : 0.f;
      float bv = p.bias ? p.bias[n] : 0.f;
#pragma unroll
      for (int r = 0; r < 8; ++r) {
        int m = row0 + ((lane >> 4) << 3) + r;
        if (m < p.M) {
          float v = acc[r] * sc + sh + bv;
          if (p.act == 1) v = fmaxf(v, 0.f);
          else if (p.act == 2) v = sigmoidf_(v);
          if (p.res) v += p.res[(long)m * p.ldres + n];
          if (p.C)   p.C[(long)m * p.ldc + n] = v;
          if (p.C16) p.C16[(long)m * p.ldc16 + n] = (_Float16)v;
        }
      }
    }
  }
}

// ---------------------------------------------------------------------------
// In-block (8 wave) GEMM and fused GRU cell, for the persistent scan kernels.
// ---------------------------------------------------------------------------
DEV void gemm_block(const Asrc16& xs, int M, int K, const _Float16* Wt, int N,
                    const float* bias, float* C, long ldc,
                    _Float16* C16, long ldc16) {
  int wid = threadIdx.x >> 5;
  int ntw = (N + 15) >> 4;
  int total = ((M + 15) >> 4) * ntw;
  for (int u = wid; u < total; u += 8) {
    int mt = u / ntw, nt = u - mt * ntw;
    int row0 = mt << 4, n0 = nt << 4;
    v8f acc = {};
    for (int k0 = 0; k0 < K; k0 += 32)
      acc = wmma_f16(frag_a16(xs, K, row0, k0), frag_b16(Wt, N, K, n0, k0), acc);
    int lane = threadIdx.x & 31;
    int n = n0 + (lane & 15);
    if (n < N) {
      float bv = bias ? bias[n] : 0.f;
#pragma unroll
      for (int r = 0; r < 8; ++r) {
        int m = row0 + ((lane >> 4) << 3) + r;
        if (m < M) {
          float v = acc[r] + bv;
          if (C)   C[(long)m * ldc + n] = v;
          if (C16) C16[(long)m * ldc16 + n] = (_Float16)v;
        }
      }
    }
  }
}

// GRU cell for batch M=32, hidden H. Each wave owns (mtile,ntile), computes the
// six gate tiles in WMMA accumulators and writes h_next directly (PyTorch GRU):
//   r = sig(ir+hr); z = sig(iz+hz); n = tanh(in + r*hn); h' = (1-z)*n + z*h
// h: f32 master (gate math); h16: f16 mirror (WMMA A operand).
DEV void gru_cell_block(const Asrc16& xs, int Kin, int H,
                        const _Float16* WihT, const _Float16* WhhT,
                        const float* bih, const float* bhh,
                        const float* h, const _Float16* h16, float* hnext) {
  int wid = threadIdx.x >> 5;
  int ntiles = H >> 4;
  Asrc16 hs; hs.p0 = h16; hs.s0 = H; hs.klen0 = 1 << 30; hs.p1 = nullptr; hs.s1 = 0;
  for (int u = wid; u < 2 * ntiles; u += 8) {
    int mt = u / ntiles, nt = u - mt * ntiles;
    int row0 = mt << 4, n0 = nt << 4;
    v8f air = {}, aiz = {}, ain = {}, ahr = {}, ahz = {}, ahn = {};
    for (int k0 = 0; k0 < Kin; k0 += 32) {
      v16h a = frag_a16(xs, Kin, row0, k0);
      air = wmma_f16(a, frag_b16(WihT + (long)0 * H * Kin, H, Kin, n0, k0), air);
      aiz = wmma_f16(a, frag_b16(WihT + (long)1 * H * Kin, H, Kin, n0, k0), aiz);
      ain = wmma_f16(a, frag_b16(WihT + (long)2 * H * Kin, H, Kin, n0, k0), ain);
    }
    for (int k0 = 0; k0 < H; k0 += 32) {
      v16h a = frag_a16(hs, H, row0, k0);
      ahr = wmma_f16(a, frag_b16(WhhT + (long)0 * H * H, H, H, n0, k0), ahr);
      ahz = wmma_f16(a, frag_b16(WhhT + (long)1 * H * H, H, H, n0, k0), ahz);
      ahn = wmma_f16(a, frag_b16(WhhT + (long)2 * H * H, H, H, n0, k0), ahn);
    }
    int lane = threadIdx.x & 31;
    int n = n0 + (lane & 15);
    float bir = bih[n], biz = bih[H + n], bin = bih[2 * H + n];
    float bhr = bhh[n], bhz = bhh[H + n], bhn = bhh[2 * H + n];
#pragma unroll
    for (int r = 0; r < 8; ++r) {
      int m = row0 + ((lane >> 4) << 3) + r;
      float rr = sigmoidf_(air[r] + bir + ahr[r] + bhr);
      float zz = sigmoidf_(aiz[r] + biz + ahz[r] + bhz);
      float nn = tanhf(ain[r] + bin + rr * (ahn[r] + bhn));
      hnext[m * H + n] = (1.f - zz) * nn + zz * h[m * H + n];
    }
  }
}

// ---------------------------------------------------------------------------
// Decoder scan: 100 teacher-forced steps, B=32. All state in LDS.
// ---------------------------------------------------------------------------
struct DecW {
  const _Float16 *aWihT, *aWhhT, *WqT, *projWT, *g1WihT, *g1WhhT,
                 *g2WihT, *g2WhhT, *melWT;
  const float *aBih, *aBhh, *vvec, *projB, *g1Bih, *g1Bhh, *g2Bih, *g2Bhh, *melB;
};

__global__ void __launch_bounds__(256) k_decoder(
    const float* __restrict__ z, const float* __restrict__ zm,
    const _Float16* __restrict__ pren16, const int* __restrict__ lens,
    DecW w, float* __restrict__ melO, _Float16* __restrict__ mel16,
    float* __restrict__ alignO) {
  extern __shared__ char smc[];
  float* ha  = (float*)smc;       // [32][256] f32 masters
  float* h1  = ha + 8192;
  float* h2  = h1 + 8192;
  float* ctx = h2 + 8192;
  float* hx  = ctx + 8192;        // h_next scratch
  float* xq  = hx + 8192;         // q / x buffer
  _Float16* ha16  = (_Float16*)(xq + 8192);   // f16 mirrors (WMMA A operands)
  _Float16* h1h   = ha16 + 8192;
  _Float16* h2h   = h1h + 8192;
  _Float16* ctx16 = h2h + 8192;
  _Float16* xq16  = ctx16 + 8192;
  float* ea = (float*)(xq16 + 8192);          // [32][180] scores
  for (int i = threadIdx.x; i < 4 * 8192; i += 256) ha[i] = 0.f;
  for (int i = threadIdx.x; i < 4 * 8192; i += 256) ha16[i] = (_Float16)0.f;
  __syncthreads();
#pragma unroll 1
  for (int t = 0; t < 100; ++t) {
    // attention GRU: input = [pren_t (128) , ctx (256)]
    Asrc16 xs; xs.p0 = pren16 + (long)t * 128; xs.s0 = 12800; xs.klen0 = 128;
    xs.p1 = ctx16; xs.s1 = 256;
    gru_cell_block(xs, 384, 256, w.aWihT, w.aWhhT, w.aBih, w.aBhh, ha, ha16, hx);
    __syncthreads();
    for (int i = threadIdx.x; i < 8192; i += 256) {
      float v = hx[i]; ha[i] = v; ha16[i] = (_Float16)v;
    }
    __syncthreads();
    // q = ha @ Wq
    Asrc16 hs; hs.p0 = ha16; hs.s0 = 256; hs.klen0 = 1 << 30; hs.p1 = nullptr; hs.s1 = 0;
    gemm_block(hs, 32, 256, w.WqT, 256, nullptr, xq, 256, nullptr, 0);
    __syncthreads();
    // additive attention scores + mask
    for (int i = threadIdx.x; i < 32 * 180; i += 256) {
      int b = i / 180, tt = i - b * 180;
      const float* zr = zm + ((long)(b * 180 + tt)) * 256;
      const float* qr = xq + b * 256;
      float e = 0.f;
      for (int d = 0; d < 256; ++d) e += tanhf(qr[d] + zr[d]) * w.vvec[d];
      ea[i] = (tt < lens[b]) ? e : -1e9f;
    }
    __syncthreads();
    if (threadIdx.x < 32) {      // softmax per batch row
      float* er = ea + threadIdx.x * 180;
      float mx = -1e30f;
      for (int tt = 0; tt < 180; ++tt) mx = fmaxf(mx, er[tt]);
      float ssum = 0.f;
      for (int tt = 0; tt < 180; ++tt) { float v = __expf(er[tt] - mx); er[tt] = v; ssum += v; }
      float inv = 1.f / ssum;
      for (int tt = 0; tt < 180; ++tt) er[tt] *= inv;
    }
    __syncthreads();
    for (int i = threadIdx.x; i < 32 * 180; i += 256) {
      int b = i / 180, tt = i - b * 180;
      alignO[(long)b * 18000 + (long)t * 180 + tt] = ea[i];
    }
    // ctx = a @ z
    for (int i = threadIdx.x; i < 8192; i += 256) {
      int b = i >> 8, d = i & 255;
      const float* zb = z + (long)b * 180 * 256 + d;
      const float* ab = ea + b * 180;
      float acc = 0.f;
      for (int tt = 0; tt < 180; ++tt) acc += ab[tt] * zb[(long)tt * 256];
      ctx[i] = acc; ctx16[i] = (_Float16)acc;
    }
    __syncthreads();
    // x = [ctx, ha] @ projW + b
    Asrc16 cs; cs.p0 = ctx16; cs.s0 = 256; cs.klen0 = 256; cs.p1 = ha16; cs.s1 = 256;
    gemm_block(cs, 32, 512, w.projWT, 256, w.projB, xq, 256, xq16, 256);
    __syncthreads();
    Asrc16 xqs; xqs.p0 = xq16; xqs.s0 = 256; xqs.klen0 = 1 << 30; xqs.p1 = nullptr; xqs.s1 = 0;
    // dec GRU1 + residual
    gru_cell_block(xqs, 256, 256, w.g1WihT, w.g1WhhT, w.g1Bih, w.g1Bhh, h1, h1h, hx);
    __syncthreads();
    for (int i = threadIdx.x; i < 8192; i += 256) {
      float v = hx[i]; h1[i] = v; h1h[i] = (_Float16)v;
    }
    __syncthreads();
    for (int i = threadIdx.x; i < 8192; i += 256) {
      float v = xq[i] + h1[i]; xq[i] = v; xq16[i] = (_Float16)v;
    }
    __syncthreads();
    // dec GRU2 + residual
    gru_cell_block(xqs, 256, 256, w.g2WihT, w.g2WhhT, w.g2Bih, w.g2Bhh, h2, h2h, hx);
    __syncthreads();
    for (int i = threadIdx.x; i < 8192; i += 256) {
      float v = hx[i]; h2[i] = v; h2h[i] = (_Float16)v;
    }
    __syncthreads();
    for (int i = threadIdx.x; i < 8192; i += 256) {
      float v = xq[i] + h2[i]; xq[i] = v; xq16[i] = (_Float16)v;
    }
    __syncthreads();
    // mel frame: [32,400] -> mel_pred[b, t*5 .. t*5+5, 80] (contiguous 400)
    gemm_block(xqs, 32, 256, w.melWT, 400, w.melB,
               melO + (long)t * 400, 40000, mel16 + (long)t * 400, 40000);
    __syncthreads();
  }
}

// ---------------------------------------------------------------------------
// Bidirectional GRU scan (CBHG): H=128, L=500, persistent block. f16 output.
// ---------------------------------------------------------------------------
struct GruW { const _Float16 *WihT, *WhhT; const float *bih, *bhh; };

__global__ void __launch_bounds__(256) k_gru(const _Float16* __restrict__ X16,
                                             GruW w, _Float16* __restrict__ out16,
                                             int reverse) {
  extern __shared__ char smc[];
  float* h  = (float*)smc;        // [32][128] f32 master
  float* hx = h + 4096;
  _Float16* h16 = (_Float16*)(hx + 4096);
  for (int i = threadIdx.x; i < 4096; i += 256) { h[i] = 0.f; h16[i] = (_Float16)0.f; }
  __syncthreads();
#pragma unroll 1
  for (int s = 0; s < 500; ++s) {
    int t = reverse ? (499 - s) : s;
    Asrc16 xs; xs.p0 = X16 + (long)t * 128; xs.s0 = 64000; xs.klen0 = 1 << 30;
    xs.p1 = nullptr; xs.s1 = 0;
    gru_cell_block(xs, 128, 128, w.WihT, w.WhhT, w.bih, w.bhh, h, h16, hx);
    __syncthreads();
    for (int i = threadIdx.x; i < 4096; i += 256) {
      float hv = hx[i];
      h[i] = hv; h16[i] = (_Float16)hv;
      int b = i >> 7, d = i & 127;
      out16[((long)(b * 500 + t)) * 128 + d] = (_Float16)hv;
    }
    __syncthreads();
  }
}

// ---------------------------------------------------------------------------
// Small utility kernels.
// ---------------------------------------------------------------------------
__global__ void k_cvt(const float* __restrict__ w, _Float16* __restrict__ wt,
                      long In, long Out) {   // f32 [In,Out] -> f16 [Out,In]
  long i = (long)blockIdx.x * blockDim.x + threadIdx.x;
  if (i < In * Out) {
    long ii = i / Out, oo = i - ii * Out;
    wt[oo * In + ii] = (_Float16)w[i];
  }
}

__global__ void k_tof16(const float* __restrict__ a, _Float16* __restrict__ b, long n) {
  long i = (long)blockIdx.x * blockDim.x + threadIdx.x;
  if (i < n) b[i] = (_Float16)a[i];
}

__global__ void k_bnfold(const float* g, const float* b, const float* m,
                         const float* v, float* sc, float* sh, int C) {
  int i = blockIdx.x * blockDim.x + threadIdx.x;
  if (i < C) {
    float s = g[i] * rsqrtf(v[i] + 1e-5f);
    sc[i] = s; sh[i] = b[i] - m[i] * s;
  }
}

__global__ void k_frames(const float* __restrict__ y, _Float16* __restrict__ fr) {
  long i = (long)blockIdx.x * blockDim.x + threadIdx.x;   // [3200,400] f16
  if (i < 1280000L) {
    long m = i / 400, j = i - m * 400;
    long b = m / 100, t = m - b * 100;
    fr[i] = (t == 0) ? (_Float16)0.f : (_Float16)y[b * 40000 + (t - 1) * 400 + j];
  }
}

__global__ void k_hw(const float* __restrict__ Hh, const float* __restrict__ Tt,
                     float* __restrict__ x, _Float16* __restrict__ x16, long n) {
  long i = (long)blockIdx.x * blockDim.x + threadIdx.x;
  if (i < n) {
    float tv = Tt[i];
    float v = Hh[i] * tv + x[i] * (1.f - tv);
    x[i] = v; x16[i] = (_Float16)v;
  }
}

// ---------------------------------------------------------------------------
// Host driver
// ---------------------------------------------------------------------------
enum {
  IN_Z = 0, IN_Y = 1, IN_LEN = 2,
  I_preW1 = 3, I_preB1, I_preW2, I_preB2,
  I_aWih = 7, I_aWhh, I_aBih, I_aBhh,
  I_Wq = 11, I_Wm, I_v,
  I_projW = 14, I_projB,
  I_g1Wih = 16, I_g1Whh, I_g1Bih, I_g1Bhh,
  I_g2Wih = 20, I_g2Whh, I_g2Bih, I_g2Bhh,
  I_melW = 24, I_melB, I_pcb,
  I_bank0 = 27,              // 8 x (W, gamma, beta, mean, var)
  I_p1W = 67, I_p1BN = 68,   // gamma,beta,mean,var
  I_p2W = 72, I_p2BN = 73,
  I_hw0 = 77,                // 4 x (Wh, bh, Wt, bt)
  I_gfWih = 93, I_gfWhh, I_gfBih, I_gfBhh,
  I_gbWih = 97, I_gbWhh, I_gbBih, I_gbBhh,
  I_linW = 101, I_linB = 102
};

extern "C" void kernel_launch(void* const* d_in, const int* in_sizes, int n_in,
                              void* d_out, int out_size, void* d_ws, size_t ws_size,
                              hipStream_t stream) {
  (void)in_sizes; (void)n_in; (void)out_size; (void)ws_size;
#define F32(i) ((const float*)d_in[(i)])
  const float* z  = F32(IN_Z);
  const float* y  = F32(IN_Y);
  const int* lens = (const int*)d_in[IN_LEN];

  float* melO = (float*)d_out;                  // [32,500,80]
  float* linO = melO + 1280000;                 // [32,500,1025]
  float* algO = linO + 16400000;                // [32,100,180]

  char* base = (char*)d_ws; size_t off = 0;
  auto aH = [&](size_t n) { _Float16* p = (_Float16*)(base + off);
                            off += (n * 2 + 255) & ~(size_t)255; return p; };
  auto aF = [&](size_t n) { float* p = (float*)(base + off);
                            off += (n * 4 + 255) & ~(size_t)255; return p; };

  // f16 transposed weights ([out][in])
  _Float16* preW1T = aH(400 * 256);
  _Float16* preW2T = aH(256 * 128);
  _Float16* aWihT  = aH(384 * 768);
  _Float16* aWhhT  = aH(256 * 768);
  _Float16* WqT    = aH(256 * 256);
  _Float16* WmT    = aH(256 * 256);
  _Float16* projWT = aH(512 * 256);
  _Float16* g1WihT = aH(256 * 768);
  _Float16* g1WhhT = aH(256 * 768);
  _Float16* g2WihT = aH(256 * 768);
  _Float16* g2WhhT = aH(256 * 768);
  _Float16* melWT  = aH(256 * 400);
  _Float16* pcbT   = aH(80 * 128);
  _Float16* bankT[8];
  for (int k = 1; k <= 8; ++k) bankT[k - 1] = aH((size_t)k * 128 * 128);
  _Float16* p1T = aH(3072 * 256);
  _Float16* p2T = aH(768 * 128);
  _Float16 *hwWhT[4], *hwWtT[4];
  for (int i = 0; i < 4; ++i) { hwWhT[i] = aH(128 * 128); hwWtT[i] = aH(128 * 128); }
  _Float16* gfWihT = aH(128 * 384);
  _Float16* gfWhhT = aH(128 * 384);
  _Float16* gbWihT = aH(128 * 384);
  _Float16* gbWhhT = aH(128 * 384);
  _Float16* linT   = aH(256 * 1025);

  // folded BN scale/shift
  float *bnS[10], *bnB[10];
  for (int i = 0; i < 8; ++i) { bnS[i] = aF(128); bnB[i] = aF(128); }
  bnS[8] = aF(256); bnB[8] = aF(256);   // proj1
  bnS[9] = aF(128); bnB[9] = aF(128);   // proj2

  // activations (f16 where only consumed as WMMA A; f32 masters elsewhere)
  _Float16* frames16 = aH(3200L * 400);
  _Float16* pren1h   = aH(3200L * 256);
  _Float16* pren16   = aH(3200L * 128);
  _Float16* z16      = aH(5760L * 256);
  float*    zmB      = aF(5760L * 256);
  _Float16* mel16    = aH(16000L * 80);
  float*    x0f      = aF(16000L * 128);   // residual (f32)
  _Float16* x016     = aH(16000L * 128);
  _Float16* bank16   = aH(16000L * 1024);
  _Float16* p1h      = aH(16000L * 256);
  float*    xh       = aF(16000L * 128);
  _Float16* xh16     = aH(16000L * 128);
  float*    hwA      = aF(16000L * 128);
  float*    hwB      = aF(16000L * 128);
  _Float16* hf16     = aH(16000L * 128);
  _Float16* hb16     = aH(16000L * 128);

  auto cvt = [&](int idx, _Float16* dst, long In, long Out) {
    long n = In * Out;
    k_cvt<<<(int)((n + 255) / 256), 256, 0, stream>>>(F32(idx), dst, In, Out);
  };
  cvt(I_preW1, preW1T, 400, 256);   cvt(I_preW2, preW2T, 256, 128);
  cvt(I_aWih, aWihT, 384, 768);     cvt(I_aWhh, aWhhT, 256, 768);
  cvt(I_Wq, WqT, 256, 256);         cvt(I_Wm, WmT, 256, 256);
  cvt(I_projW, projWT, 512, 256);
  cvt(I_g1Wih, g1WihT, 256, 768);   cvt(I_g1Whh, g1WhhT, 256, 768);
  cvt(I_g2Wih, g2WihT, 256, 768);   cvt(I_g2Whh, g2WhhT, 256, 768);
  cvt(I_melW, melWT, 256, 400);     cvt(I_pcb, pcbT, 80, 128);
  for (int k = 1; k <= 8; ++k) cvt(I_bank0 + 5 * (k - 1), bankT[k - 1], (long)k * 128, 128);
  cvt(I_p1W, p1T, 3072, 256);       cvt(I_p2W, p2T, 768, 128);
  for (int i = 0; i < 4; ++i) {
    cvt(I_hw0 + 4 * i + 0, hwWhT[i], 128, 128);
    cvt(I_hw0 + 4 * i + 2, hwWtT[i], 128, 128);
  }
  cvt(I_gfWih, gfWihT, 128, 384);   cvt(I_gfWhh, gfWhhT, 128, 384);
  cvt(I_gbWih, gbWihT, 128, 384);   cvt(I_gbWhh, gbWhhT, 128, 384);
  cvt(I_linW, linT, 256, 1025);

  auto bnfold = [&](int gidx, float* s, float* sh, int C) {
    k_bnfold<<<(C + 255) / 256, 256, 0, stream>>>(F32(gidx), F32(gidx + 1),
                                                  F32(gidx + 2), F32(gidx + 3), s, sh, C);
  };
  for (int k = 0; k < 8; ++k) bnfold(I_bank0 + 5 * k + 1, bnS[k], bnB[k], 128);
  bnfold(I_p1BN, bnS[8], bnB[8], 256);
  bnfold(I_p2BN, bnS[9], bnB[9], 128);

  auto gemmN = [&](const _Float16* A, long lda, const _Float16* Wt, const float* bias,
                   float* C, long ldc, _Float16* C16, long ldc16,
                   int M, int N, int K, int act) {
    GemmP p{}; p.A = A; p.lda = lda; p.Wt = Wt; p.bias = bias;
    p.C = C; p.ldc = ldc; p.C16 = C16; p.ldc16 = ldc16;
    p.M = M; p.N = N; p.K = K; p.act = act;
    int tiles = ((M + 15) / 16) * ((N + 15) / 16);
    k_gemm<0><<<(tiles + 3) / 4, 128, 0, stream>>>(p);
  };
  auto gemmConv = [&](const _Float16* X, int Cin, int kw, const _Float16* Wt,
                      const float* sc, const float* sh, const float* res, long ldres,
                      float* C, long ldc, _Float16* C16, long ldc16,
                      int N, int act, bool pool) {
    GemmP p{}; p.X = X; p.Lc = 500; p.Cin = Cin; p.kw = kw; p.pad = kw / 2;
    p.Wt = Wt; p.scale = sc; p.shift = sh; p.res = res; p.ldres = ldres;
    p.C = C; p.ldc = ldc; p.C16 = C16; p.ldc16 = ldc16;
    p.M = 16000; p.N = N; p.K = kw * Cin; p.act = act;
    int tiles = 1000 * ((N + 15) / 16);
    if (pool) k_gemm<2><<<(tiles + 3) / 4, 128, 0, stream>>>(p);
    else      k_gemm<1><<<(tiles + 3) / 4, 128, 0, stream>>>(p);
  };

  // ---- prenet + memory projection ----
  k_frames<<<(1280000 + 255) / 256, 256, 0, stream>>>(y, frames16);
  k_tof16<<<(int)((5760L * 256 + 255) / 256), 256, 0, stream>>>(z, z16, 5760L * 256);
  gemmN(frames16, 400, preW1T, F32(I_preB1), nullptr, 0, pren1h, 256, 3200, 256, 400, 1);
  gemmN(pren1h, 256, preW2T, F32(I_preB2), nullptr, 0, pren16, 128, 3200, 128, 256, 1);
  gemmN(z16, 256, WmT, nullptr, zmB, 256, nullptr, 0, 5760, 256, 256, 0);

  // ---- decoder scan (mel + alignments) ----
  DecW dw;
  dw.aWihT = aWihT; dw.aWhhT = aWhhT; dw.WqT = WqT; dw.projWT = projWT;
  dw.g1WihT = g1WihT; dw.g1WhhT = g1WhhT; dw.g2WihT = g2WihT; dw.g2WhhT = g2WhhT;
  dw.melWT = melWT;
  dw.aBih = F32(I_aBih); dw.aBhh = F32(I_aBhh); dw.vvec = F32(I_v);
  dw.projB = F32(I_projB);
  dw.g1Bih = F32(I_g1Bih); dw.g1Bhh = F32(I_g1Bhh);
  dw.g2Bih = F32(I_g2Bih); dw.g2Bhh = F32(I_g2Bhh);
  dw.melB = F32(I_melB);
  size_t decSmem = 6 * 8192 * 4 + 5 * 8192 * 2 + 32 * 180 * 4;  // ~295 KB < 320 KB
  k_decoder<<<1, 256, decSmem, stream>>>(z, zmB, pren16, lens, dw, melO, mel16, algO);

  // ---- CBHG postnet ----
  gemmN(mel16, 80, pcbT, nullptr, x0f, 128, x016, 128, 16000, 128, 80, 0);
  for (int k = 1; k <= 8; ++k)
    gemmConv(x016, 128, k, bankT[k - 1], bnS[k - 1], bnB[k - 1], nullptr, 0,
             nullptr, 0, bank16 + (k - 1) * 128, 1024, 128, 1, false);
  gemmConv(bank16, 1024, 3, p1T, bnS[8], bnB[8], nullptr, 0,
           nullptr, 0, p1h, 256, 256, 1, true);
  gemmConv(p1h, 256, 3, p2T, bnS[9], bnB[9], x0f, 128,
           xh, 128, xh16, 128, 128, 0, false);
  for (int i = 0; i < 4; ++i) {
    gemmN(xh16, 128, hwWhT[i], F32(I_hw0 + 4 * i + 1), hwA, 128, nullptr, 0,
          16000, 128, 128, 1);
    gemmN(xh16, 128, hwWtT[i], F32(I_hw0 + 4 * i + 3), hwB, 128, nullptr, 0,
          16000, 128, 128, 2);
    k_hw<<<(2048000 + 255) / 256, 256, 0, stream>>>(hwA, hwB, xh, xh16, 2048000L);
  }
  GruW gwf{gfWihT, gfWhhT, F32(I_gfBih), F32(I_gfBhh)};
  GruW gwb{gbWihT, gbWhhT, F32(I_gbBih), F32(I_gbBhh)};
  size_t gruSmem = 2 * 4096 * 4 + 4096 * 2;
  k_gru<<<1, 256, gruSmem, stream>>>(xh16, gwf, hf16, 0);
  k_gru<<<1, 256, gruSmem, stream>>>(xh16, gwb, hb16, 1);
  {
    GemmP p{}; p.A = hf16; p.lda = 128; p.A2 = hb16; p.lda2 = 128; p.ksplit = 128;
    p.Wt = linT; p.bias = F32(I_linB); p.C = linO; p.ldc = 1025;
    p.M = 16000; p.N = 1025; p.K = 256; p.act = 0;
    int tiles = 1000 * 65;
    k_gemm<3><<<(tiles + 3) / 4, 128, 0, stream>>>(p);
  }
#undef F32
}
